// MT_MaxViT_25159918420305
// MI455X (gfx1250) — matrix operand order repaired and checked by reading, not compiled, not run on hardware
//
#include <hip/hip_runtime.h>
#include <hip/hip_bf16.h>

typedef __attribute__((ext_vector_type(16))) __bf16 v16bf;
typedef __attribute__((ext_vector_type(8)))  float  v8f;

#define DIM   512
#define HEADS 16
#define NTOK  64
#define HD    32
#define CPBH  256
#define NWAVE 4
#define HPW   (HEADS / NWAVE)

static __device__ __forceinline__ v8f vzero8() {
  v8f z;
#pragma unroll
  for (int i = 0; i < 8; ++i) z[i] = 0.f;
  return z;
}

static __device__ __forceinline__ v8f wmma_bf16(v16bf a, v16bf b, v8f c) {
  // D = A(16x32 bf16) * B(32x16 bf16) + C(16x16 f32)
  return __builtin_amdgcn_wmma_f32_16x16x32_bf16(false, a, false, b, (short)0, c, false, false);
}

// A fragment (16x32 bf16) from row-major f32 source. Lane L: row m0+L%16,
// K = kb*32 + (L/16)*16 .. +15 (contiguous).
static __device__ __forceinline__ v16bf load_a_f32(const float* __restrict__ base,
                                                   int ld, int m0, int kb, int lane) {
  const int m = m0 + (lane & 15);
  const int hl = lane >> 4;
  const float* p = base + (size_t)m * ld + kb * 32 + hl * 16;
  v16bf a;
#pragma unroll
  for (int t = 0; t < 8; ++t) {
    a[t]     = (__bf16)p[t];
    a[t + 8] = (__bf16)p[t + 8];
  }
  return a;
}

// A fragment from row-major bf16 source (LDS or global).
static __device__ __forceinline__ v16bf load_a_bf16(const __bf16* base,
                                                    int ld, int m0, int kb, int lane) {
  const int m = m0 + (lane & 15);
  const int hl = lane >> 4;
  const __bf16* p = base + (size_t)m * ld + kb * 32 + hl * 16;
  v16bf a;
#pragma unroll
  for (int t = 0; t < 8; ++t) {
    a[t]     = p[t];
    a[t + 8] = p[t + 8];
  }
  return a;
}

// B fragment from band-transposed packed weights: wp[kb][n][s] = W[kb*32+s][n].
// Lane L: column n0+L%16, K = kb*32 + (L/16)*16 .. +15 -> 32B contiguous.
static __device__ __forceinline__ v16bf load_b_packed(const __bf16* __restrict__ wp,
                                                      int ncols, int kb, int n0, int lane) {
  const __bf16* p = wp + ((size_t)(kb * ncols + n0 + (lane & 15)) << 5) + ((lane >> 4) << 4);
  return *(const v16bf*)p;
}

// B fragment where B[k][n] with 'base' stored row-major over n (ld elems per row),
// k contiguous within a row (used for q^T: base = q_s[token][dim]).
static __device__ __forceinline__ v16bf load_b_rowmajor16(const __bf16* base,
                                                          int ld, int n0, int k0, int lane) {
  const __bf16* p = base + (size_t)(n0 + (lane & 15)) * ld + k0 + ((lane >> 4) << 4);
  return *(const v16bf*)p;
}

// 64x32 projection slice: dst = src(64x512 f32) @ Wpacked[:, ncol0:ncol0+32] + bias, *mul.
// Stores bf16 to LDS either row-major [m][d] or transposed [d][m].
static __device__ __forceinline__ void gemm_headslice(
    const float* __restrict__ src, const __bf16* __restrict__ wp, int wncols, int ncol0,
    const float* __restrict__ bias_base, float mul, __bf16* dst, bool transposeStore, int lane) {
  const int hl = lane >> 4, lm = lane & 15;
  v8f acc[4][2];
#pragma unroll
  for (int mt = 0; mt < 4; ++mt)
#pragma unroll
    for (int nt = 0; nt < 2; ++nt) acc[mt][nt] = vzero8();

  for (int kb = 0; kb < 16; ++kb) {
    v16bf a[4], bw[2];
#pragma unroll
    for (int mt = 0; mt < 4; ++mt) a[mt] = load_a_f32(src, DIM, mt * 16, kb, lane);
#pragma unroll
    for (int nt = 0; nt < 2; ++nt) bw[nt] = load_b_packed(wp, wncols, kb, ncol0 + nt * 16, lane);
    if (kb + 1 < 16) {
      const __bf16* pf = wp + ((size_t)((kb + 1) * wncols + ncol0 + lm) << 5);
      __builtin_prefetch(pf, 0, 1);  // global_prefetch_b8 into caches
    }
#pragma unroll
    for (int mt = 0; mt < 4; ++mt)
#pragma unroll
      for (int nt = 0; nt < 2; ++nt) acc[mt][nt] = wmma_bf16(a[mt], bw[nt], acc[mt][nt]);
  }

#pragma unroll
  for (int nt = 0; nt < 2; ++nt) {
    const float bv = bias_base[nt * 16 + lm];
#pragma unroll
    for (int mt = 0; mt < 4; ++mt)
#pragma unroll
      for (int r = 0; r < 8; ++r) {
        const float v = (acc[mt][nt][r] + bv) * mul;
        const int m = mt * 16 + hl * 8 + r;
        const int d = nt * 16 + lm;
        if (transposeStore) dst[d * NTOK + m] = (__bf16)v;
        else                dst[m * HD + d]   = (__bf16)v;
      }
  }
}

__global__ __launch_bounds__(NWAVE * 32, 1) void fused_window_attn(
    const float* __restrict__ x, const float* __restrict__ KV,
    const float* __restrict__ q_b, const float* __restrict__ kv_b,
    const float* __restrict__ proj_b, const float* __restrict__ tau,
    const __bf16* __restrict__ qw_p, const __bf16* __restrict__ kvw_p,
    const __bf16* __restrict__ pw_p, const float* __restrict__ biasw,
    __bf16* __restrict__ ostage, float* __restrict__ out) {
  const int b = blockIdx.x;
  const int lane = threadIdx.x & 31;
  const int wave = threadIdx.x >> 5;
  const int hl = lane >> 4;
  const int lm = lane & 15;

  __shared__ __align__(32) __bf16 sh_q[NWAVE][NTOK * HD];
  __shared__ __align__(32) __bf16 sh_k[NWAVE][NTOK * HD];
  __shared__ __align__(32) __bf16 sh_vT[NWAVE][HD * NTOK];
  __shared__ float sh_qn[NWAVE][NTOK];
  __shared__ float sh_kn[NWAVE][NTOK];

  __bf16* q_s  = sh_q[wave];
  __bf16* k_s  = sh_k[wave];
  __bf16* vT_s = sh_vT[wave];
  float*  qn_s = sh_qn[wave];
  float*  kn_s = sh_kn[wave];

  const float* xb  = x  + (size_t)b * NTOK * DIM;
  const float* kvb = KV + (size_t)b * NTOK * DIM;
  const float scale = 0.17677669529663687f;  // HD^-0.5
  __bf16* ob = ostage + (size_t)b * NTOK * DIM;

  for (int hi = 0; hi < HPW; ++hi) {
    const int h = wave * HPW + hi;

    // ---- Q, K, V head slices via WMMA; Q scaled; V stored transposed ----
    gemm_headslice(xb,  qw_p,  DIM,     h * HD,       q_b  + h * HD,       scale, q_s,  false, lane);
    gemm_headslice(kvb, kvw_p, 2 * DIM, h * HD,       kv_b + h * HD,       1.0f,  k_s,  false, lane);
    gemm_headslice(kvb, kvw_p, 2 * DIM, DIM + h * HD, kv_b + DIM + h * HD, 1.0f,  vT_s, true,  lane);
    __builtin_amdgcn_wave_barrier();

    // ---- row norms |q|, |k| (wave-private LDS, in-order DS ops) ----
    for (int m = lane; m < NTOK; m += 32) {
      float sq = 0.f, sk = 0.f;
#pragma unroll
      for (int d = 0; d < HD; ++d) {
        const float qv = (float)q_s[m * HD + d];
        const float kv = (float)k_s[m * HD + d];
        sq += qv * qv;
        sk += kv * kv;
      }
      qn_s[m] = sqrtf(sq);
      kn_s[m] = sqrtf(sk);
    }
    __builtin_amdgcn_wave_barrier();

    // ---- attn^T = K (64x32) @ Q^T (32x64): softmax axis lands on M dim ----
    v8f at[4][4];
#pragma unroll
    for (int mt = 0; mt < 4; ++mt)
#pragma unroll
      for (int nt = 0; nt < 4; ++nt) at[mt][nt] = vzero8();
    {
      v16bf ak[4], bq[4];
#pragma unroll
      for (int mt = 0; mt < 4; ++mt) ak[mt] = load_a_bf16(k_s, HD, mt * 16, 0, lane);
#pragma unroll
      for (int nt = 0; nt < 4; ++nt) bq[nt] = load_b_rowmajor16(q_s, HD, nt * 16, 0, lane);
#pragma unroll
      for (int mt = 0; mt < 4; ++mt)
#pragma unroll
        for (int nt = 0; nt < 4; ++nt) at[mt][nt] = wmma_bf16(ak[mt], bq[nt], at[mt][nt]);
    }

    // ---- cosine norm, tau clamp, CPB bias, softmax over k (registers only) ----
    const float* taub  = tau   + (size_t)h * NTOK * NTOK;
    const float* biasb = biasw + (size_t)h * NTOK * NTOK;
#pragma unroll
    for (int nt = 0; nt < 4; ++nt) {
      const int qq = nt * 16 + lm;
      const float qnv = qn_s[qq];
      float mx = -3.0e38f;
#pragma unroll
      for (int mt = 0; mt < 4; ++mt)
#pragma unroll
        for (int r = 0; r < 8; ++r) {
          const int kk = mt * 16 + hl * 8 + r;
          const float dn = fmaxf(qnv * kn_s[kk], 1e-6f);
          const float tv = fmaxf(taub[qq * NTOK + kk], 0.01f);
          const float v = at[mt][nt][r] / dn / tv + biasb[qq * NTOK + kk];
          at[mt][nt][r] = v;
          mx = fmaxf(mx, v);
        }
      mx = fmaxf(mx, __shfl_xor(mx, 16));
      float sum = 0.f;
#pragma unroll
      for (int mt = 0; mt < 4; ++mt)
#pragma unroll
        for (int r = 0; r < 8; ++r) {
          const float e = __expf(at[mt][nt][r] - mx);
          at[mt][nt][r] = e;
          sum += e;
        }
      sum += __shfl_xor(sum, 16);
      const float inv = 1.f / sum;
#pragma unroll
      for (int mt = 0; mt < 4; ++mt)
#pragma unroll
        for (int r = 0; r < 8; ++r) at[mt][nt][r] *= inv;
    }

    // ---- out^T = V^T (32x64) @ attn^T (64x64); B built from regs via xor-16 shuffles ----
    v8f oc[2][4];
#pragma unroll
    for (int mt2 = 0; mt2 < 2; ++mt2)
#pragma unroll
      for (int nt = 0; nt < 4; ++nt) oc[mt2][nt] = vzero8();
#pragma unroll
    for (int kt = 0; kt < 2; ++kt) {
      v16bf av[2];
#pragma unroll
      for (int mt2 = 0; mt2 < 2; ++mt2) av[mt2] = load_a_bf16(vT_s, NTOK, mt2 * 16, kt, lane);
#pragma unroll
      for (int nt = 0; nt < 4; ++nt) {
        v16bf bb;
#pragma unroll
        for (int r = 0; r < 8; ++r) {
          const float a0 = at[kt * 2 + 0][nt][r];
          const float a1 = at[kt * 2 + 1][nt][r];
          const float s0 = __shfl_xor(a0, 16);
          const float s1 = __shfl_xor(a1, 16);
          bb[r]     = (__bf16)(hl ? s1 : a0);
          bb[r + 8] = (__bf16)(hl ? a1 : s0);
        }
#pragma unroll
        for (int mt2 = 0; mt2 < 2; ++mt2) oc[mt2][nt] = wmma_bf16(av[mt2], bb, oc[mt2][nt]);
      }
    }

    // ---- stage per-head output (bf16) [token][h*32+d] ----
#pragma unroll
    for (int mt2 = 0; mt2 < 2; ++mt2)
#pragma unroll
      for (int nt = 0; nt < 4; ++nt)
#pragma unroll
        for (int r = 0; r < 8; ++r) {
          const int q = nt * 16 + lm;
          const int d = mt2 * 16 + hl * 8 + r;
          ob[(size_t)q * DIM + h * HD + d] = (__bf16)oc[mt2][nt][r];
        }
  }

  __threadfence();
  __syncthreads();

  // ---- final projection: (64x512 bf16) @ proj_w (512x512) + proj_b -> f32 out ----
  const __bf16* Ab = ob;
  for (int c = 0; c < 2; ++c) {
    const int n0 = wave * 128 + c * 64;
    v8f pc[4][4];
#pragma unroll
    for (int mt = 0; mt < 4; ++mt)
#pragma unroll
      for (int nt = 0; nt < 4; ++nt) pc[mt][nt] = vzero8();
    for (int kb = 0; kb < 16; ++kb) {
      v16bf a[4], bw[4];
#pragma unroll
      for (int mt = 0; mt < 4; ++mt) a[mt] = load_a_bf16(Ab, DIM, mt * 16, kb, lane);
#pragma unroll
      for (int nt = 0; nt < 4; ++nt) bw[nt] = load_b_packed(pw_p, DIM, kb, n0 + nt * 16, lane);
#pragma unroll
      for (int mt = 0; mt < 4; ++mt)
#pragma unroll
        for (int nt = 0; nt < 4; ++nt) pc[mt][nt] = wmma_bf16(a[mt], bw[nt], pc[mt][nt]);
    }
#pragma unroll
    for (int nt = 0; nt < 4; ++nt) {
      const float pb = proj_b[n0 + nt * 16 + lm];
#pragma unroll
      for (int mt = 0; mt < 4; ++mt)
#pragma unroll
        for (int r = 0; r < 8; ++r) {
          const int m = mt * 16 + hl * 8 + r;
          out[((size_t)b * NTOK + m) * DIM + n0 + nt * 16 + lm] = pc[mt][nt][r] + pb;
        }
    }
  }
}

// Band-transpose W (K x ncols, f32) into WMMA-B packed bf16: wp[(k/32)*ncols+n][k%32].
__global__ void pack_weight(const float* __restrict__ w, __bf16* __restrict__ wp,
                            int K, int ncols) {
  const int idx = blockIdx.x * blockDim.x + threadIdx.x;
  if (idx >= K * ncols) return;
  const int k = idx / ncols;
  const int n = idx % ncols;
  wp[((size_t)((k >> 5) * ncols + n) << 5) + (k & 31)] = (__bf16)w[idx];
}

static __device__ __forceinline__ float signed_log1p(float d) {
  const float l = log1pf(fabsf(d));
  return d < 0.f ? -l : l;
}

// CPB MLP: bias[h][i][j] = relu(logrel(i,j) @ w1 + b1) @ w2 + b2 (batch-independent).
__global__ void cpb_bias_kernel(const float* __restrict__ w1, const float* __restrict__ b1,
                                const float* __restrict__ w2, const float* __restrict__ b2,
                                float* __restrict__ biasw) {
  const int i = blockIdx.x;   // query token (64)
  const int j = threadIdx.x;  // key token (64)
  const float r0 = signed_log1p((float)((i >> 3) - (j >> 3)));
  const float r1 = signed_log1p((float)((i & 7) - (j & 7)));
  float acc[HEADS];
#pragma unroll
  for (int hh = 0; hh < HEADS; ++hh) acc[hh] = b2[hh];
  for (int c = 0; c < CPBH; ++c) {
    float hv = r0 * w1[c] + r1 * w1[CPBH + c] + b1[c];
    hv = fmaxf(hv, 0.f);
#pragma unroll
    for (int hh = 0; hh < HEADS; ++hh) acc[hh] += hv * w2[c * HEADS + hh];
  }
#pragma unroll
  for (int hh = 0; hh < HEADS; ++hh)
    biasw[((size_t)hh * NTOK + i) * NTOK + j] = acc[hh];
}

extern "C" void kernel_launch(void* const* d_in, const int* in_sizes, int n_in,
                              void* d_out, int out_size, void* d_ws, size_t ws_size,
                              hipStream_t stream) {
  const float* x      = (const float*)d_in[0];
  const float* KV     = (const float*)d_in[1];
  const float* q_w    = (const float*)d_in[2];
  const float* q_b    = (const float*)d_in[3];
  const float* kv_w   = (const float*)d_in[4];
  const float* kv_b   = (const float*)d_in[5];
  const float* proj_w = (const float*)d_in[6];
  const float* proj_b = (const float*)d_in[7];
  const float* cpb_w1 = (const float*)d_in[8];
  const float* cpb_b1 = (const float*)d_in[9];
  const float* cpb_w2 = (const float*)d_in[10];
  const float* cpb_b2 = (const float*)d_in[11];
  const float* tau    = (const float*)d_in[12];
  float* out = (float*)d_out;

  char* ws = (char*)d_ws;
  // ws layout (all 32B-aligned):
  float*  biasw  = (float*)(ws);                                        // 16*64*64*4   = 256 KB
  __bf16* qw_p   = (__bf16*)(ws + (262144));                            // 512*512*2    = 512 KB
  __bf16* kvw_p  = (__bf16*)(ws + (262144 + 524288));                   // 512*1024*2   = 1 MB
  __bf16* pw_p   = (__bf16*)(ws + (262144 + 524288 + 1048576));         // 512*512*2    = 512 KB
  __bf16* ostage = (__bf16*)(ws + (262144 + 524288 + 1048576 + 524288)); // 1024*64*512*2 = 64 MB

  pack_weight<<<(512 * 512 + 255) / 256, 256, 0, stream>>>(q_w, qw_p, 512, 512);
  pack_weight<<<(512 * 1024 + 255) / 256, 256, 0, stream>>>(kv_w, kvw_p, 512, 1024);
  pack_weight<<<(512 * 512 + 255) / 256, 256, 0, stream>>>(proj_w, pw_p, 512, 512);
  cpb_bias_kernel<<<64, 64, 0, stream>>>(cpb_w1, cpb_b1, cpb_w2, cpb_b2, biasw);

  fused_window_attn<<<1024, NWAVE * 32, 0, stream>>>(
      x, KV, q_b, kv_b, proj_b, tau, qw_p, kvw_p, pw_p, biasw, ostage, out);
}
